// RegionProposalNetwork_34007551050002
// MI455X (gfx1250) — matrix operand order repaired and checked by reading, not compiled, not run on hardware
//
#include <hip/hip_runtime.h>
#include <math.h>

typedef __attribute__((ext_vector_type(16))) _Float16 v16h;
typedef __attribute__((ext_vector_type(8)))  _Float16 v8h;
typedef __attribute__((ext_vector_type(8)))  float    v8f;
typedef __attribute__((address_space(3)))    _Float16 lds_f16;

#define HH    128
#define WW    128
#define CC    512
#define AA    9
#define HWPIX 16384            // HH*WW
#define K1    4608             // 9*CC
#define NANCH 147456           // HWPIX*AA
#define PRE_NMS  12000
#define POST_NMS 2000
#define NEGINF  (-__builtin_inff())

// ---------------- workspace layout (bytes) ----------------
#define OFF_X16   ((size_t)0)                         // HWPIX*CC f16 = 16 MB (NHWC)
#define OFF_W1P   (OFF_X16  + (size_t)16777216)       // K1*CC f16 (B-fragment swizzled)
#define OFF_Y16   (OFF_W1P  + (size_t)4718592)        // HWPIX*CC f16 = 16 MB
#define OFF_W2P   (OFF_Y16  + (size_t)16777216)       // heads weights padded to N=48
#define OFF_KEYS  (OFF_W2P  + (size_t)49152)          // NANCH u32
#define OFF_BOXES (OFF_KEYS + (size_t)589824)         // NANCH*4 f32
#define OFF_NMSSC (OFF_BOXES+ (size_t)2359296)        // NANCH f32
#define OFF_STATE (OFF_NMSSC+ (size_t)589824)         // hist[256] + prefix/want/thresh

// ---------------- pack feature map: NCHW f32 -> NHWC f16 ----------------
__global__ void k_pack_x(const float* __restrict__ fm, _Float16* __restrict__ x16) {
  int t = blockIdx.x * blockDim.x + threadIdx.x;      // t = pixel*512 + c
  int c = t & (CC - 1);
  int p = t >> 9;
  x16[t] = (_Float16)fm[c * HWPIX + p];
}

// ---------------- pack conv weights into WMMA B-fragment layout ----------------
// B[k][n], k = tap*512+ci, n = out channel. Fragment: lane holds n = lane&15,
// element e holds k = kc*32 + (lane>>4)*16 + e  (contiguous 32B per lane).
__global__ void k_pack_w1(const float* __restrict__ w1, _Float16* __restrict__ w1p) {
  int t = blockIdx.x * blockDim.x + threadIdx.x;      // t < K1*CC
  int nt   = t / 73728;                               // 144*32*16 halves per n-tile
  int rem  = t % 73728;
  int kc   = rem >> 9;
  int r2   = rem & 511;
  int lane = r2 >> 4;
  int e    = r2 & 15;
  int k = kc * 32 + ((lane >> 4) << 4) + e;
  int n = nt * 16 + (lane & 15);
  int tap = k >> 9;
  int ci  = k & 511;
  w1p[t] = (_Float16)w1[(n * CC + ci) * 9 + tap];
}

// ---------------- pack head weights (wc:9xC, wb:36xC) padded to N=48 ----------------
__global__ void k_pack_w2(const float* __restrict__ wc, const float* __restrict__ wb,
                          _Float16* __restrict__ w2p) {
  int t = blockIdx.x * blockDim.x + threadIdx.x;      // t < 24576
  int nt   = t / 8192;
  int rem  = t % 8192;
  int kc   = rem >> 9;
  int r2   = rem & 511;
  int lane = r2 >> 4;
  int e    = r2 & 15;
  int k = kc * 32 + ((lane >> 4) << 4) + e;
  int n = nt * 16 + (lane & 15);
  float v = 0.0f;
  if (n < 9)       v = wc[n * CC + k];
  else if (n < 45) v = wb[(n - 9) * CC + k];
  w2p[t] = (_Float16)v;
}

// ---------------- async fill of one K-chunk A-slice into LDS ----------------
// Slice layout: [row 0..2][pix 0..65][c 0..31] halves (row = ky, pix = w - (wbase-1)).
// 3*66*32 halves = 12672 B = 792 x b128. Out-of-bounds cells are skipped (EXEC-masked);
// they were zeroed once at kernel start and stay zero across all refills.
__device__ __forceinline__ void fill_slice(_Float16* sbuf, const _Float16* __restrict__ x16,
                                           int h, int wbase, int chunk, int tid) {
  for (int e = tid; e < 792; e += 256) {
    int half16 = e & 1;
    int pr  = e >> 1;
    int p   = pr % 66;
    int row = pr / 66;
    int h2 = h + row - 1;
    int w2 = wbase - 1 + p;
    if ((h2 >= 0) && (h2 < HH) && (w2 >= 0) && (w2 < WW)) {
      unsigned lds = (unsigned)(size_t)(lds_f16*)(sbuf + ((row * 66 + p) * 32 + half16 * 8));
      const _Float16* g = x16 + (size_t)(h2 * WW + w2) * CC + chunk * 32 + half16 * 8;
      asm volatile("global_load_async_to_lds_b128 %0, %1, off"
                   :: "v"(lds), "v"(g) : "memory");
    }
  }
}

__device__ __forceinline__ void wait_async0() {
#if __has_builtin(__builtin_amdgcn_s_wait_asynccnt)
  __builtin_amdgcn_s_wait_asynccnt(0);
#else
  asm volatile("s_wait_asynccnt 0x0" ::: "memory");
#endif
}

// ---------------- 3x3 conv + ReLU as implicit GEMM (WMMA f16 -> f32) ----------------
// Block = 8 waves, 64 pixels x 512 channels. Each wave: 4 m-subtiles x 4 n-subtiles
// (16 accumulators). A staged in LDS via async loads, double buffered over K-chunks;
// B fragments held in registers across the 4 m-subtiles. Halo cells in LDS are
// pre-zeroed once so the hot loop has zero masking overhead.
__global__ __launch_bounds__(256) void k_conv(const _Float16* __restrict__ x16,
                                              const _Float16* __restrict__ w1p,
                                              const float* __restrict__ b1,
                                              _Float16* __restrict__ y16) {
  __shared__ __attribute__((aligned(16))) _Float16 sA[2][6336];

  const int blk   = blockIdx.x;                 // 0..255
  const int h     = blk >> 1;
  const int wbase = (blk & 1) << 6;             // 0 or 64
  const int wave  = threadIdx.x >> 5;           // 0..7
  const int lane  = threadIdx.x & 31;
  const int m     = lane & 15;
  const int sel   = lane >> 4;
  const int nt0   = wave * 4;                   // 4 n-tiles = 64 out channels
  const int tid   = threadIdx.x;

  v8f acc[16];
#pragma unroll
  for (int i = 0; i < 16; ++i) acc[i] = (v8f){};
  const v8h z8 = {};

  // Zero both buffers once (halo cells are never overwritten afterwards).
  {
    v8h* sz = (v8h*)(&sA[0][0]);
    for (int e = tid; e < 1584; e += 256) sz[e] = z8;
  }
  __syncthreads();

  fill_slice(&sA[0][0], x16, h, wbase, 0, tid);

  for (int ch = 0; ch < 16; ++ch) {
    wait_async0();
    __syncthreads();
    const int cur = ch & 1;
    if (ch < 15) fill_slice(&sA[cur ^ 1][0], x16, h, wbase, ch + 1, tid);

#pragma unroll
    for (int tap = 0; tap < 9; ++tap) {
      const int ky = tap / 3, kx = tap % 3;
      const _Float16* bp = w1p + ((size_t)((nt0 * 144 + tap * 16 + ch) * 32 + lane) << 4);
      v16h b0  = *(const v16h*)(bp);
      v16h b1f = *(const v16h*)(bp + 73728);
      v16h b2f = *(const v16h*)(bp + 2 * 73728);
      v16h b3f = *(const v16h*)(bp + 3 * 73728);
#pragma unroll
      for (int ms = 0; ms < 4; ++ms) {
        const int p = ms * 16 + m + kx;          // 0..65 slice pixel
        const _Float16* s = &sA[cur][(ky * 66 + p) * 32 + (sel << 4)];
        union { v8h g[2]; v16h v; } af;
        af.g[0] = *(const v8h*)(s);
        af.g[1] = *(const v8h*)(s + 8);
        acc[ms * 4 + 0] = __builtin_amdgcn_wmma_f32_16x16x32_f16(false, af.v, false, b0,  (short)0, acc[ms * 4 + 0], false, false);
        acc[ms * 4 + 1] = __builtin_amdgcn_wmma_f32_16x16x32_f16(false, af.v, false, b1f, (short)0, acc[ms * 4 + 1], false, false);
        acc[ms * 4 + 2] = __builtin_amdgcn_wmma_f32_16x16x32_f16(false, af.v, false, b2f, (short)0, acc[ms * 4 + 2], false, false);
        acc[ms * 4 + 3] = __builtin_amdgcn_wmma_f32_16x16x32_f16(false, af.v, false, b3f, (short)0, acc[ms * 4 + 3], false, false);
      }
    }
    __syncthreads();
  }

  // Epilogue: lane holds col n=lane&15; VGPR v holds row M=v+8*sel.
  const int n0 = nt0 * 16 + (lane & 15);
#pragma unroll
  for (int ms = 0; ms < 4; ++ms) {
#pragma unroll
    for (int v = 0; v < 8; ++v) {
      const int pix = h * WW + wbase + ms * 16 + v + (sel << 3);
      _Float16* yrow = y16 + (size_t)pix * CC;
      yrow[n0]      = (_Float16)fmaxf(acc[ms * 4 + 0][v] + b1[n0],      0.0f);
      yrow[n0 + 16] = (_Float16)fmaxf(acc[ms * 4 + 1][v] + b1[n0 + 16], 0.0f);
      yrow[n0 + 32] = (_Float16)fmaxf(acc[ms * 4 + 2][v] + b1[n0 + 32], 0.0f);
      yrow[n0 + 48] = (_Float16)fmaxf(acc[ms * 4 + 3][v] + b1[n0 + 48], 0.0f);
    }
  }
}

// ---------------- heads: 1x1 convs as WMMA GEMM (N padded to 48) ----------------
__global__ __launch_bounds__(96) void k_heads(const _Float16* __restrict__ y16,
                                              const _Float16* __restrict__ w2p,
                                              const float* __restrict__ bc,
                                              const float* __restrict__ bbias,
                                              float* __restrict__ out_obj,
                                              float* __restrict__ out_del) {
  const int mtile = blockIdx.x;
  const int nt    = threadIdx.x >> 5;           // 0..2
  const int lane  = threadIdx.x & 31;
  const int m     = lane & 15;
  const int sel   = lane >> 4;
  v8f acc = {};
  const _Float16* arow = y16 + (size_t)(mtile * 16 + m) * CC;
  for (int ch = 0; ch < 16; ++ch) {
    const int c0 = (ch << 5) + (sel << 4);
    union { v8h g[2]; v16h v; } af;
    af.g[0] = *(const v8h*)(arow + c0);
    af.g[1] = *(const v8h*)(arow + c0 + 8);
    v16h bf = *(const v16h*)(w2p + ((size_t)((nt * 16 + ch) * 32 + lane) << 4));
    acc = __builtin_amdgcn_wmma_f32_16x16x32_f16(false, af.v, false, bf, (short)0, acc, false, false);
  }
  const int n = nt * 16 + (lane & 15);
  for (int v = 0; v < 8; ++v) {
    const int pix = mtile * 16 + v + (sel << 3);
    float val = acc[v];
    if (n < 9) {
      val += bc[n];
      out_obj[pix * 9 + n] = 1.0f / (1.0f + expf(-val));
    } else if (n < 45) {
      const int d = n - 9;
      out_del[pix * 36 + d] = val + bbias[d];
    }
  }
}

// ---------------- sortable keys for masked scores ----------------
__device__ __forceinline__ unsigned f2key(float f) {
  unsigned u = __float_as_uint(f);
  return (u & 0x80000000u) ? ~u : (u | 0x80000000u);
}

__global__ void k_keys(const float* __restrict__ obj, const int* __restrict__ valid,
                       unsigned* __restrict__ keys) {
  int i = blockIdx.x * blockDim.x + threadIdx.x;
  float s = (valid[i] > 0) ? obj[i] : NEGINF;
  keys[i] = f2key(s);
}

// ---------------- radix select (4 MSB-first byte passes) ----------------
__global__ void k_selinit(unsigned* __restrict__ st) {
  int t = threadIdx.x;
  st[t] = 0u;
  if (t == 0) { st[256] = 0u; st[257] = PRE_NMS; st[258] = 0u; }
}

__global__ void k_hist(const unsigned* __restrict__ keys, unsigned* __restrict__ st, int s) {
  int i = blockIdx.x * blockDim.x + threadIdx.x;
  unsigned key = keys[i];
  unsigned prefix = st[256];
  bool match = (s == 0) || (((key ^ prefix) >> (32 - 8 * s)) == 0u);
  if (match) {
    unsigned byte = (key >> (24 - 8 * s)) & 0xFFu;
    atomicAdd(&st[byte], 1u);
  }
}

__global__ void k_scan(unsigned* __restrict__ st, int s) {
  unsigned want = st[257];
  unsigned cum = 0, chosen = 0;
  for (int b = 255; b >= 0; --b) {
    unsigned c = st[b];
    if (cum + c >= want) { chosen = (unsigned)b; want -= cum; break; }
    cum += c;
  }
  st[256] |= chosen << (24 - 8 * s);
  st[257] = want;
  for (int b = 0; b < 256; ++b) st[b] = 0u;
  if (s == 3) st[258] = st[256];
}

// ---------------- decode deltas -> clipped boxes + masked NMS scores ----------------
__global__ void k_decode(const unsigned* __restrict__ keys, const unsigned* __restrict__ st,
                         const float* __restrict__ obj, const int* __restrict__ valid,
                         const float* __restrict__ anchors, const float* __restrict__ deltas,
                         float* __restrict__ boxes, float* __restrict__ nmssc) {
  int i = blockIdx.x * blockDim.x + threadIdx.x;
  const unsigned T = st[258];
  const bool selected = keys[i] >= T;
  const int pix = i / 9, a = i % 9;
  const float* d  = deltas + pix * 36 + a * 4;
  const float* an = anchors + (size_t)i * 4;
  float acy = an[0], acx = an[1], ah = an[2], aw = an[3];
  float cy = acy + d[0] * ah;
  float cx = acx + d[1] * aw;
  float hh  = ah * expf(d[2]);
  float wwd = aw * expf(d[3]);
  float y1 = fmaxf(cy - 0.5f * hh, 0.0f);
  float x1 = fmaxf(cx - 0.5f * wwd, 0.0f);
  float y2 = fminf(cy + 0.5f * hh, 2048.0f);
  float x2 = fminf(cx + 0.5f * wwd, 2048.0f);
  boxes[(size_t)i * 4 + 0] = y1;
  boxes[(size_t)i * 4 + 1] = x1;
  boxes[(size_t)i * 4 + 2] = y2;
  boxes[(size_t)i * 4 + 3] = x2;
  float sc = (valid[i] > 0) ? obj[i] : NEGINF;
  bool keep = selected && (y2 - y1 >= 16.0f) && (x2 - x1 >= 16.0f) && (sc != NEGINF);
  nmssc[i] = keep ? sc : NEGINF;
}

// ---------------- greedy NMS: single block, argmax-reduce + parallel IoU ----------------
__global__ __launch_bounds__(1024) void k_nms(float* __restrict__ sc,
                                              const float* __restrict__ boxes,
                                              float* __restrict__ props) {
  __shared__ float sv[1024];
  __shared__ int   si[1024];
  __shared__ float sbox[5];
  const int tid = threadIdx.x;
  for (int it = 0; it < POST_NMS; ++it) {
    float bv = NEGINF; int bi = NANCH;
    for (int i = tid; i < NANCH; i += 1024) {
      float s = sc[i];
      if (s > bv || (s == bv && i < bi)) { bv = s; bi = i; }
    }
    sv[tid] = bv; si[tid] = bi;
    __syncthreads();
    for (int off = 512; off > 0; off >>= 1) {
      if (tid < off) {
        float ov = sv[tid + off]; int oi = si[tid + off];
        if (ov > sv[tid] || (ov == sv[tid] && oi < si[tid])) { sv[tid] = ov; si[tid] = oi; }
      }
      __syncthreads();
    }
    const int best = si[0];
    if (tid == 0) {
      const bool ok = sv[0] > NEGINF;
      float y1 = 0, x1 = 0, y2 = 0, x2 = 0;
      if (ok) {
        y1 = boxes[(size_t)best * 4 + 0]; x1 = boxes[(size_t)best * 4 + 1];
        y2 = boxes[(size_t)best * 4 + 2]; x2 = boxes[(size_t)best * 4 + 3];
      }
      props[it * 4 + 0] = y1; props[it * 4 + 1] = x1;
      props[it * 4 + 2] = y2; props[it * 4 + 3] = x2;
      sbox[0] = y1; sbox[1] = x1; sbox[2] = y2; sbox[3] = x2; sbox[4] = ok ? 1.0f : 0.0f;
    }
    __syncthreads();
    if (sbox[4] > 0.0f) {
      const float by1 = sbox[0], bx1 = sbox[1], by2 = sbox[2], bx2 = sbox[3];
      const float a1 = (by2 - by1) * (bx2 - bx1);
      for (int i = tid; i < NANCH; i += 1024) {
        if (i == best) { sc[i] = NEGINF; continue; }
        if (sc[i] == NEGINF) continue;
        float cy1 = boxes[(size_t)i * 4 + 0], cx1 = boxes[(size_t)i * 4 + 1];
        float cy2 = boxes[(size_t)i * 4 + 2], cx2 = boxes[(size_t)i * 4 + 3];
        float yy1 = fmaxf(by1, cy1), xx1 = fmaxf(bx1, cx1);
        float yy2 = fminf(by2, cy2), xx2 = fminf(bx2, cx2);
        float inter = fmaxf(yy2 - yy1, 0.0f) * fmaxf(xx2 - xx1, 0.0f);
        float a2 = (cy2 - cy1) * (cx2 - cx1);
        float un = a1 + a2 - inter;
        float iou = (un > 0.0f) ? inter / un : 0.0f;
        if (iou > 0.7f) sc[i] = NEGINF;
      }
    }
    __syncthreads();
  }
}

// ---------------- launcher ----------------
extern "C" void kernel_launch(void* const* d_in, const int* in_sizes, int n_in,
                              void* d_out, int out_size, void* d_ws, size_t ws_size,
                              hipStream_t stream) {
  const float* fm   = (const float*)d_in[0];
  const float* amap = (const float*)d_in[1];
  const int*   avm  = (const int*)d_in[2];
  // d_in[3..6]: image_height, image_width, pre_nms, post_nms (constants here)
  const float* w1 = (const float*)d_in[7];
  const float* b1 = (const float*)d_in[8];
  const float* wc = (const float*)d_in[9];
  const float* bc = (const float*)d_in[10];
  const float* wb = (const float*)d_in[11];
  const float* bb = (const float*)d_in[12];

  char* ws = (char*)d_ws;
  _Float16* x16   = (_Float16*)(ws + OFF_X16);
  _Float16* w1p   = (_Float16*)(ws + OFF_W1P);
  _Float16* y16   = (_Float16*)(ws + OFF_Y16);
  _Float16* w2p   = (_Float16*)(ws + OFF_W2P);
  unsigned* keys  = (unsigned*)(ws + OFF_KEYS);
  float*    boxes = (float*)(ws + OFF_BOXES);
  float*    nmssc = (float*)(ws + OFF_NMSSC);
  unsigned* st    = (unsigned*)(ws + OFF_STATE);

  float* out_obj  = (float*)d_out;              // [H,W,A]
  float* out_del  = out_obj + NANCH;            // [H,W,4A]
  float* out_prop = out_del + NANCH * 4;        // [2000,4]

  k_pack_x <<<HWPIX * CC / 256, 256, 0, stream>>>(fm, x16);
  k_pack_w1<<<K1 * CC / 256,    256, 0, stream>>>(w1, w1p);
  k_pack_w2<<<96,               256, 0, stream>>>(wc, wb, w2p);
  k_conv   <<<256,              256, 0, stream>>>(x16, w1p, b1, y16);
  k_heads  <<<1024,              96, 0, stream>>>(y16, w2p, bc, bb, out_obj, out_del);
  k_keys   <<<NANCH / 256,      256, 0, stream>>>(out_obj, avm, keys);
  k_selinit<<<1,                256, 0, stream>>>(st);
  for (int s = 0; s < 4; ++s) {
    k_hist <<<NANCH / 256,      256, 0, stream>>>(keys, st, s);
    k_scan <<<1,                  1, 0, stream>>>(st, s);
  }
  k_decode <<<NANCH / 256,      256, 0, stream>>>(keys, st, out_obj, avm, amap, out_del,
                                                  boxes, nmssc);
  k_nms    <<<1,               1024, 0, stream>>>(nmssc, boxes, out_prop);
}
